// Attention_21174188770080
// MI455X (gfx1250) — matrix operand order repaired and checked by reading, not compiled, not run on hardware
//
#include <hip/hip_runtime.h>
#include <cstdint>

#define DEV __device__ __forceinline__

typedef __bf16 bf16;
typedef bf16  bf16x8  __attribute__((ext_vector_type(8)));
typedef bf16  bf16x16 __attribute__((ext_vector_type(16)));
typedef float f32x8   __attribute__((ext_vector_type(8)));
typedef uint32_t u32x4 __attribute__((ext_vector_type(4)));
typedef uint32_t u32x8 __attribute__((ext_vector_type(8)));

namespace {
constexpr int NB = 4, L = 1024, D = 768, H = 12, HD = 64;
constexpr int BL = NB * L;           // 4096
constexpr float SCALE = 0.125f;      // 1/sqrt(64)
}

#define IDX16 0,1,2,3,4,5,6,7,8,9,10,11,12,13,14,15

// A fragment: 16x32 bf16 tile, row-major, base = tile origin, lda in elements.
// ISA 7.12.2 layout: lanes 0-15 row M=lane, halves K=0..7 then 16..23;
// lanes 16-31 same rows, K=8..15 then 24..31.
DEV bf16x16 load_a_frag(const bf16* __restrict__ A, int lda, int lane) {
  const int half = lane >> 4;
  const int r    = lane & 15;
  const bf16* p  = A + (size_t)r * lda + half * 8;
  bf16x8 lo = *reinterpret_cast<const bf16x8*>(p);
  bf16x8 hi = *reinterpret_cast<const bf16x8*>(p + 16);
  return __builtin_shufflevector(lo, hi, IDX16);
}

// B fragment: 32x16 bf16 tile (KxN), stored transposed so column n is a
// contiguous K-run. lanes 0-15 hold K=0..7 then 16..23, lanes 16-31 K=8..15 then 24..31.
DEV bf16x16 load_b_frag(const bf16* __restrict__ Bt, int ldk, int lane) {
  const int half = lane >> 4;
  const int n    = lane & 15;
  const bf16* p  = Bt + (size_t)n * ldk + half * 8;
  bf16x8 lo = *reinterpret_cast<const bf16x8*>(p);
  bf16x8 hi = *reinterpret_cast<const bf16x8*>(p + 16);
  return __builtin_shufflevector(lo, hi, IDX16);
}

DEV f32x8 wmma_bf16(bf16x16 a, bf16x16 b, f32x8 c) {
  return __builtin_amdgcn_wmma_f32_16x16x32_bf16(
      false, a, false, b, (short)0, c, false, false);
}

// ---------------- conversion kernels ----------------
__global__ void k_f32_to_bf16(const float* __restrict__ in,
                              bf16* __restrict__ out, int n) {
  int i = blockIdx.x * blockDim.x + threadIdx.x;
  if (i < n) out[i] = (bf16)in[i];
}

// W [rows, cols] f32 -> Wt [cols, rows] bf16
__global__ void k_transpose_bf16(const float* __restrict__ W,
                                 bf16* __restrict__ Wt, int rows, int cols) {
  int i = blockIdx.x * blockDim.x + threadIdx.x;
  if (i < rows * cols) {
    int k = i / cols, n = i % cols;
    Wt[(size_t)n * rows + k] = (bf16)W[i];
  }
}

// ---------------- QKV projection (TDM-staged A panel) ----------------
// grid = (256 M-tiles, 3 weights); block = 12 waves, wave w = head/64-col strip.
// The 16x768 bf16 A panel is DMA'd to LDS once per block by the Tensor Data
// Mover; all waves read A fragments from LDS (ds_load_b128).
// Q,K stored head-split [B,H,L,HD]; V stored transposed [B,H,HD,L].
__global__ void k_qkv(const bf16* __restrict__ xb,
                      const bf16* __restrict__ Wqt,
                      const bf16* __restrict__ Wkt,
                      const bf16* __restrict__ Wvt,
                      bf16* __restrict__ Qh, bf16* __restrict__ Kh,
                      bf16* __restrict__ Vt) {
  __shared__ __align__(16) bf16 sA[16 * D];   // 24 KB
  const int lane  = threadIdx.x & 31;
  const int w     = threadIdx.x >> 5;         // wave id == head id
  const int mt    = blockIdx.x;
  const int which = blockIdx.y;               // 0=Q,1=K,2=V

  if (threadIdx.x < 32) {
    // 2D Tensor DMA descriptor: tile = 768 x 16 bf16 panel of xb.
    uint64_t ga = (uint64_t)(uintptr_t)xb + (uint64_t)mt * 16 * D * 2;
    uint32_t ldsoff = (uint32_t)(uintptr_t)(&sA[0]);
    u32x4 g0 = { 1u,                                   // count=1, user mode
                 ldsoff,                               // lds_addr
                 (uint32_t)ga,                         // global_addr[31:0]
                 (uint32_t)((ga >> 32) & 0x01FFFFFFu) | 0x80000000u }; // type=2
    u32x8 g1 = { 0x00010000u,                          // mask=0, data_size=2B
                 (uint32_t)D << 16,                    // tensor_dim0 = 768
                 16u << 16,                            // tensor_dim1 = 16 rows
                 (uint32_t)D << 16,                    // tile_dim0 = 768
                 16u,                                  // tile_dim1 = 16
                 (uint32_t)D,                          // tensor_dim0_stride
                 0u, 0u };
    asm volatile("tensor_load_to_lds %0, %1" :: "s"(g0), "s"(g1) : "memory");
    __builtin_amdgcn_s_wait_tensorcnt(0);
  }
  __syncthreads();

  const bf16* Wt = (which == 0) ? Wqt : (which == 1) ? Wkt : Wvt;
  const int half = lane >> 4;
  const int r    = lane & 15;
  const int n    = lane & 15;

  f32x8 acc[4] = {};
#pragma unroll 2
  for (int k = 0; k < D; k += 32) {
    bf16x8 alo = *reinterpret_cast<const bf16x8*>(&sA[r * D + k + half * 8]);
    bf16x8 ahi = *reinterpret_cast<const bf16x8*>(&sA[r * D + k + 16 + half * 8]);
    bf16x16 a = __builtin_shufflevector(alo, ahi, IDX16);
#pragma unroll
    for (int dt = 0; dt < 4; ++dt) {
      const bf16* Bb = Wt + (size_t)(w * 64 + dt * 16) * D + k;
      acc[dt] = wmma_bf16(a, load_b_frag(Bb, D, lane), acc[dt]);
    }
  }

#pragma unroll
  for (int dt = 0; dt < 4; ++dt) {
    const int d = dt * 16 + n;     // head-local column, head h == w
#pragma unroll
    for (int tt = 0; tt < 8; ++tt) {
      int m  = mt * 16 + half * 8 + tt;
      int b_ = m >> 10;
      int l  = m & (L - 1);
      bf16 v = (bf16)acc[dt][tt];
      if (which == 0)      Qh[(((size_t)b_ * H + w) * L + l) * HD + d] = v;
      else if (which == 1) Kh[(((size_t)b_ * H + w) * L + l) * HD + d] = v;
      else                 Vt[(((size_t)b_ * H + w) * HD + d) * L + l] = v;
    }
  }
}

// ---------------- scores + pre-softmax talking-heads mix ----------------
__global__ void k_scores_premix(const bf16* __restrict__ Qh,
                                const bf16* __restrict__ Kh,
                                const float* __restrict__ pre,
                                float* __restrict__ S) {
  __shared__ float sS[H][16][17];
  const int lane = threadIdx.x & 31;
  const int h    = threadIdx.x >> 5;
  const int it = blockIdx.x, jt = blockIdx.y, b = blockIdx.z;
  const bf16* Ab = Qh + (((size_t)b * H + h) * L + it * 16) * HD;
  const bf16* Bb = Kh + (((size_t)b * H + h) * L + jt * 16) * HD;
  f32x8 acc = {};
  acc = wmma_bf16(load_a_frag(Ab, HD, lane),      load_b_frag(Bb, HD, lane),      acc);
  acc = wmma_bf16(load_a_frag(Ab + 32, HD, lane), load_b_frag(Bb + 32, HD, lane), acc);
  const int half = lane >> 4, n = lane & 15;
#pragma unroll
  for (int tt = 0; tt < 8; ++tt) sS[h][half * 8 + tt][n] = acc[tt] * SCALE;
  __syncthreads();
  float pg[H];
#pragma unroll
  for (int hh = 0; hh < H; ++hh) pg[hh] = pre[hh * H + h];   // g = h (wave id)
  float* out = S + (((size_t)b * H + h) * L + it * 16) * L + jt * 16;
#pragma unroll
  for (int tt = 0; tt < 8; ++tt) {
    int m = half * 8 + tt;
    float s = 0.f;
#pragma unroll
    for (int hh = 0; hh < H; ++hh) s += sS[hh][m][n] * pg[hh];
    out[(size_t)m * L + n] = s;
  }
}

// ---------------- softmax + post-softmax talking-heads mix ----------------
__global__ void k_softmax_postmix(const float* __restrict__ S,
                                  const float* __restrict__ post,
                                  bf16* __restrict__ P) {
  __shared__ float sE[H][L];   // 48 KB
  __shared__ float sR[H];
  const int lane = threadIdx.x & 31;
  const int h    = threadIdx.x >> 5;
  const int i = blockIdx.x, b = blockIdx.y;
  const float* row = S + (((size_t)b * H + h) * L + i) * L;
  float v[L / 32];
  float mx = -3.402823466e38f;
#pragma unroll
  for (int j = 0; j < L / 32; ++j) {
    v[j] = row[j * 32 + lane];
    mx = fmaxf(mx, v[j]);
  }
#pragma unroll
  for (int off = 16; off >= 1; off >>= 1) mx = fmaxf(mx, __shfl_xor(mx, off, 32));
  float sum = 0.f;
#pragma unroll
  for (int j = 0; j < L / 32; ++j) {
    float e = __expf(v[j] - mx);
    sE[h][j * 32 + lane] = e;
    sum += e;
  }
#pragma unroll
  for (int off = 16; off >= 1; off >>= 1) sum += __shfl_xor(sum, off, 32);
  if (lane == 0) sR[h] = 1.0f / sum;
  __syncthreads();
  float pg[H];
#pragma unroll
  for (int hh = 0; hh < H; ++hh) pg[hh] = post[hh * H + h] * sR[hh];
  bf16* out = P + (((size_t)b * H + h) * L + i) * L;
  for (int j = lane; j < L; j += 32) {
    float s = 0.f;
#pragma unroll
    for (int hh = 0; hh < H; ++hh) s += sE[hh][j] * pg[hh];
    out[j] = (bf16)s;
  }
}

// ---------------- AV GEMM, 16x64 strip per wave, reshape on store ----------
__global__ void k_av(const bf16* __restrict__ P, const bf16* __restrict__ Vt,
                     bf16* __restrict__ AVr) {
  const int lane = threadIdx.x & 31;
  const int wave = threadIdx.x >> 5;
  const int gw = blockIdx.x * 8 + wave;   // [0, 4*12*64)
  const int bg = gw >> 6;
  const int b = bg / H, g = bg % H;
  const int it = gw & 63;
  const bf16* Ab = P  + (((size_t)b * H + g) * L + it * 16) * L;
  const bf16* Bb = Vt + (((size_t)b * H + g) * HD) * L;
  f32x8 acc[4] = {};
#pragma unroll 2
  for (int k = 0; k < L; k += 32) {
    bf16x16 a = load_a_frag(Ab + k, L, lane);
#pragma unroll
    for (int dt = 0; dt < 4; ++dt)
      acc[dt] = wmma_bf16(a, load_b_frag(Bb + (size_t)dt * 16 * L + k, L, lane), acc[dt]);
  }
  const int half = lane >> 4;
#pragma unroll
  for (int dt = 0; dt < 4; ++dt) {
    const int d = dt * 16 + (lane & 15);
#pragma unroll
    for (int tt = 0; tt < 8; ++tt) {
      int i  = it * 16 + half * 8 + tt;
      int r  = g * L + i;              // flat (h,l) row within batch
      int l2 = r / H, h2 = r % H;      // raw reshape -> (l2, h2)
      AVr[((size_t)b * L + l2) * D + h2 * HD + d] = (bf16)acc[dt][tt];
    }
  }
}

// ---------------- output GEMM + bias, 16x64 strip per wave ----------------
__global__ void k_out(const bf16* __restrict__ AVr, const bf16* __restrict__ Wot,
                      const float* __restrict__ bo, float* __restrict__ out) {
  const int lane = threadIdx.x & 31;
  const int wave = threadIdx.x >> 5;
  const int gw = blockIdx.x * 8 + wave;   // [0, 256*12)
  const int mt = gw / 12, ng = gw % 12;
  const bf16* Ab = AVr + (size_t)(mt * 16) * D;
  const bf16* Bb = Wot + (size_t)(ng * 64) * D;
  f32x8 acc[4] = {};
#pragma unroll 2
  for (int k = 0; k < D; k += 32) {
    bf16x16 a = load_a_frag(Ab + k, D, lane);
#pragma unroll
    for (int dt = 0; dt < 4; ++dt)
      acc[dt] = wmma_bf16(a, load_b_frag(Bb + (size_t)dt * 16 * D + k, D, lane), acc[dt]);
  }
  const int half = lane >> 4;
#pragma unroll
  for (int dt = 0; dt < 4; ++dt) {
    const int col = ng * 64 + dt * 16 + (lane & 15);
    float bias = bo[col];
#pragma unroll
    for (int tt = 0; tt < 8; ++tt) {
      int m = mt * 16 + half * 8 + tt;
      out[(size_t)m * D + col] = acc[dt][tt] + bias;
    }
  }
}

extern "C" void kernel_launch(void* const* d_in, const int* in_sizes, int n_in,
                              void* d_out, int out_size, void* d_ws, size_t ws_size,
                              hipStream_t stream) {
  const float* x    = (const float*)d_in[0];
  const float* Wq   = (const float*)d_in[1];
  const float* Wk   = (const float*)d_in[2];
  const float* Wv   = (const float*)d_in[3];
  const float* pre  = (const float*)d_in[4];
  const float* post = (const float*)d_in[5];
  const float* Wo   = (const float*)d_in[6];
  const float* bo   = (const float*)d_in[7];
  float* out = (float*)d_out;

  char* p = (char*)d_ws;
  auto alloc = [&](size_t bytes) {
    char* r = p;
    p += (bytes + 255) & ~size_t(255);
    return r;
  };
  bf16*  xb  = (bf16*)alloc((size_t)BL * D * 2);
  bf16*  Wqt = (bf16*)alloc((size_t)D * D * 2);
  bf16*  Wkt = (bf16*)alloc((size_t)D * D * 2);
  bf16*  Wvt = (bf16*)alloc((size_t)D * D * 2);
  bf16*  Wot = (bf16*)alloc((size_t)D * D * 2);
  bf16*  Qh  = (bf16*)alloc((size_t)NB * H * L * HD * 2);
  bf16*  Kh  = (bf16*)alloc((size_t)NB * H * L * HD * 2);
  bf16*  Vt  = (bf16*)alloc((size_t)NB * H * HD * L * 2);
  float* S   = (float*)alloc((size_t)NB * H * L * L * 4);
  bf16*  P   = (bf16*)alloc((size_t)NB * H * L * L * 2);
  bf16*  AVr = (bf16*)alloc((size_t)BL * D * 2);

  { int n = BL * D;
    k_f32_to_bf16<<<(n + 255) / 256, 256, 0, stream>>>(x, xb, n); }
  { int n = D * D, nb = (n + 255) / 256;
    k_transpose_bf16<<<nb, 256, 0, stream>>>(Wq, Wqt, D, D);
    k_transpose_bf16<<<nb, 256, 0, stream>>>(Wk, Wkt, D, D);
    k_transpose_bf16<<<nb, 256, 0, stream>>>(Wv, Wvt, D, D);
    k_transpose_bf16<<<nb, 256, 0, stream>>>(Wo, Wot, D, D); }

  dim3 gq(BL / 16, 3);                       // 256 M-tiles x {Q,K,V}
  k_qkv<<<gq, H * 32, 0, stream>>>(xb, Wqt, Wkt, Wvt, Qh, Kh, Vt);

  dim3 g2(L / 16, L / 16, NB);
  k_scores_premix<<<g2, H * 32, 0, stream>>>(Qh, Kh, pre, S);

  dim3 g3(L, NB);
  k_softmax_postmix<<<g3, H * 32, 0, stream>>>(S, post, P);

  k_av<<<384, 256, 0, stream>>>(P, Vt, AVr);     // 3072 waves
  k_out<<<384, 256, 0, stream>>>(AVr, Wot, bo, out);  // 3072 waves
}
